// ParallelHyenaOperator_38774964748499
// MI455X (gfx1250) — matrix operand order repaired and checked by reading, not compiled, not run on hardware
//
#include <hip/hip_runtime.h>
#include <stdint.h>

#define D_MODEL 768
#define B_SIZE  2
#define SEQLEN  8192
#define THREADS 256

typedef __bf16 bf16_t;
typedef bf16_t v16bf __attribute__((ext_vector_type(16)));
typedef float  v8f   __attribute__((ext_vector_type(8)));
typedef unsigned int u32v4 __attribute__((ext_vector_type(4)));
typedef int          i32v4 __attribute__((ext_vector_type(4)));
typedef int          i32v8 __attribute__((ext_vector_type(8)));

union Frag {
  v16bf v;
  unsigned short s[16];
  u32v4 q[2];
};

__device__ __forceinline__ unsigned short f32_to_bf16_rne(float f) {
  unsigned int x = __float_as_uint(f);
  x += 0x7FFFu + ((x >> 16) & 1u);          // round-to-nearest-even
  return (unsigned short)(x >> 16);
}

// ---------------------------------------------------------------------------
// Transposed blocked-Toeplitz matmul:  Y^T = U^T * T^T
//   D[m,n] = sum_c A[m,c]*B[c,n],  m = tile-local block index, n = p (pos in
//   block), c = (delta in {0,1}, q):  c = q + 16*delta.
//   A[m,c] = u[16*(I - d0 - delta) + q]       (I = 16*t + m)
//   B[c,n] = k[16*(d0 + delta) + n - q]
// Operand layouts used here (wave32; A/B in contiguous k order, not AMD's table order):
//   A: lane = M (&15), halves h -> K = h + 16*(lane>=16)
//      => per lane one contiguous 16-half run at block (I-d0-lhi),
//         byte offset 32*(blk-lhi) and +16  -> two aligned ds_load_b128.
//   B: lanes 0-15: N=lane, K=h; lanes 16-31: N=lane-16, K=16+h
//      => k[16*d0 + 16*lhi + N - h]: descending taps.  The filter is stored
//         REVERSED in LDS (lds_kr[L-1-i] = k[i]) so this becomes an ascending
//         contiguous read -> two ds_load_b128, no per-half reversal movs.
//   C/D: lane N = lane&15, M = r + 8*(lane>=16)  -> coalesced epilogue.
// ---------------------------------------------------------------------------

template <bool KEDGE>
__device__ __forceinline__ void build_k(Frag& kf, const unsigned short* lds_kr,
                                        int d0, int lmask, int lhi) {
  const int base = (d0 << 4) + (lhi << 4) + lmask;   // highest tap index
#pragma unroll
  for (int hh = 0; hh < 16; ++hh) {
    if (KEDGE) {
      const int ki = base - hh;                      // filter tap index
      const int ii = ki < 0 ? (SEQLEN - 1) : (SEQLEN - 1 - ki);
      unsigned short kv = lds_kr[ii];
      kf.s[hh] = (ki < 0) ? (unsigned short)0 : kv;  // s==0 triangular edge
    } else {
      kf.s[hh] = lds_kr[(SEQLEN - 1 - base) + hh];   // contiguous ascending
    }
  }
}

template <bool EDGE>
__device__ __forceinline__ void load_a(Frag& a, const unsigned short* lds_u,
                                       int t, int d0, int lmask, int lhi) {
  const int blk = 16 * t + lmask - d0;      // lane-half block = blk - lhi (16 contiguous)
  if (EDGE) {
    const int b0 = blk - lhi, b1 = b0;
    const int c0 = b0 < 0 ? 0 : b0;
    const int c1 = b1 < 0 ? 0 : b1;
    a.q[0] = *(const u32v4*)&lds_u[(c0 << 4)];
    a.q[1] = *(const u32v4*)&lds_u[(c1 << 4) + 8];
    const u32v4 z = {0u, 0u, 0u, 0u};
    if (b0 < 0) a.q[0] = z;                 // v_cndmask, no EXEC change
    if (b1 < 0) a.q[1] = z;
  } else {
    a.q[0] = *(const u32v4*)&lds_u[((blk - lhi) << 4)];
    a.q[1] = *(const u32v4*)&lds_u[((blk - lhi) << 4) + 8];
  }
}

// One phase: s in [sb, se], tiles J0..3 active; EDGE selects only on tile J0,
// KEDGE (filter edge) only for the peeled s==0 call.
template <int J0, bool EDGE, bool KEDGE>
__device__ __forceinline__ void phase(int sb, int se, int wave, int lmask, int lhi,
                                      const unsigned short* lds_kr,
                                      const unsigned short* lds_u, v8f* acc) {
  for (int s = sb; s <= se; ++s) {
    const int d0 = 2 * s;
    Frag kf;
    build_k<KEDGE>(kf, lds_kr, d0, lmask, lhi);
    Frag a[4];
#pragma unroll
    for (int j = J0; j < 4; ++j) {          // load all A frags first, then MMA
      const int t = wave + 8 * j;
      if (EDGE && j == J0) load_a<true>(a[j], lds_u, t, d0, lmask, lhi);
      else                 load_a<false>(a[j], lds_u, t, d0, lmask, lhi);
    }
#pragma unroll
    for (int j = J0; j < 4; ++j) {
      acc[j] = __builtin_amdgcn_wmma_f32_16x16x32_bf16(
          false, a[j].v, false, kf.v, (short)0, acc[j], false, false);
    }
  }
}

__global__ __launch_bounds__(THREADS, 1)
void hyena_toeplitz_wmma(const float* __restrict__ x1,
                         const float* __restrict__ x2,
                         const float* __restrict__ v,
                         const float* __restrict__ hflt,
                         const float* __restrict__ dbias,
                         float* __restrict__ out) {
  __shared__ __align__(16) float          lds_h[SEQLEN];   // 32 KB TDM landing
  __shared__ __align__(16) unsigned short lds_kr[SEQLEN];  // 16 KB reversed bf16 filter
  __shared__ __align__(16) unsigned short lds_u[SEQLEN];   // 16 KB bf16 x2*v

  const int chan = (int)blockIdx.x;          // 0 .. B*D-1
  const int d    = chan % D_MODEL;
  const size_t ch = (size_t)chan * SEQLEN;

  // ---- Tensor Data Mover: DMA h[d, 0:8192] (fp32, 32 KB) into LDS ----
  if (threadIdx.x < 32) {
    unsigned long long ga = (unsigned long long)(uintptr_t)(hflt + (size_t)d * SEQLEN);
    unsigned lds_addr = (unsigned)(uintptr_t)(void*)lds_h;
    u32v4 g0;
    g0[0] = 1u;                                              // count=1, user mode
    g0[1] = lds_addr;
    g0[2] = (unsigned)(ga & 0xFFFFFFFFull);
    g0[3] = (unsigned)((ga >> 32) & 0x1FFFFFFull) | 0x80000000u; // addr hi | type=2
    i32v8 g1;
    g1[0] = (2 << 16);                       // data_size = 4B
    g1[1] = (SEQLEN & 0xFFFF) << 16;         // tensor_dim0[15:0]
    g1[2] = (SEQLEN >> 16) | (1 << 16);      // tensor_dim0 hi | tensor_dim1 = 1
    g1[3] = (SEQLEN & 0xFFFF) << 16;         // tile_dim0 = 8192
    g1[4] = 1;                               // tile_dim1 = 1
    g1[5] = SEQLEN;                          // tensor_dim0_stride
    g1[6] = 0;
    g1[7] = 0;
    i32v4 g2 = {0, 0, 0, 0};
    i32v4 g3 = {0, 0, 0, 0};
    i32v8 g4 = {0, 0, 0, 0, 0, 0, 0, 0};
    __builtin_amdgcn_tensor_load_to_lds(g0, g1, g2, g3, g4, 0);
    __builtin_amdgcn_s_wait_tensorcnt(0);
  }
  __syncthreads();

  // ---- Phase 1: k = h*exp(-r_d*t) stored reversed, u = x2*v -> bf16 ----
  {
    const float r_d  = __expf(2.302585093f * (2.0f * (float)d * (1.0f / (float)(D_MODEL - 1))));
    const float tscl = 1.0f / (float)(SEQLEN - 1);
    for (int i = (int)threadIdx.x; i < SEQLEN; i += THREADS) {
      float dec = __expf(-r_d * ((float)i * tscl));
      lds_kr[(SEQLEN - 1) - i] = f32_to_bf16_rne(lds_h[i] * dec);
      lds_u[i] = f32_to_bf16_rne(x2[ch + i] * v[ch + i]);
    }
    __builtin_prefetch(&x1[ch + (size_t)threadIdx.x * 32], 0, 1);
  }
  __syncthreads();

  const int wave  = (int)threadIdx.x >> 5;
  const int ln    = (int)threadIdx.x & 31;
  const int lmask = ln & 15;
  const int lhi   = ln >> 4;

  v8f acc[4] = {};

  const int t0 = wave, t1 = wave + 8, t2 = wave + 16, t3 = wave + 24;
  const unsigned short* lk = lds_kr;
  const unsigned short* lu = lds_u;

  // s = 0: filter triangular edge + tile0 boundary
  phase<0, true,  true >(0, 0, wave, lmask, lhi, lk, lu, acc);
  // steady / edge pairs per retiring tile (tile t: steady s<8t, edge [8t,8t+7])
  phase<0, false, false>(1,              8 * t0 - 1, wave, lmask, lhi, lk, lu, acc);
  phase<0, true,  false>(8 * t0 < 1 ? 1 : 8 * t0, 8 * t0 + 7, wave, lmask, lhi, lk, lu, acc);
  phase<1, false, false>(8 * t0 + 8,     8 * t1 - 1, wave, lmask, lhi, lk, lu, acc);
  phase<1, true,  false>(8 * t1,         8 * t1 + 7, wave, lmask, lhi, lk, lu, acc);
  phase<2, false, false>(8 * t1 + 8,     8 * t2 - 1, wave, lmask, lhi, lk, lu, acc);
  phase<2, true,  false>(8 * t2,         8 * t2 + 7, wave, lmask, lhi, lk, lu, acc);
  phase<3, false, false>(8 * t2 + 8,     8 * t3 - 1, wave, lmask, lhi, lk, lu, acc);
  phase<3, true,  false>(8 * t3,         8 * t3 + 7, wave, lmask, lhi, lk, lu, acc);

  // ---- Epilogue: out = (y + u*d_bias) * x1, coalesced (l contiguous in lane)
  const float db = dbias[d];
#pragma unroll
  for (int j = 0; j < 4; ++j) {
    const int t = wave + 8 * j;
#pragma unroll
    for (int r = 0; r < 8; ++r) {
      const int m = r + (lhi << 3);          // D layout: M = r + 8*(lane>=16)
      const int l = 256 * t + 16 * m + lmask;
      const float uu = x2[ch + l] * v[ch + l];
      out[ch + l] = (acc[j][r] + uu * db) * x1[ch + l];
    }
  }
}

extern "C" void kernel_launch(void* const* d_in, const int* in_sizes, int n_in,
                              void* d_out, int out_size, void* d_ws, size_t ws_size,
                              hipStream_t stream) {
  (void)in_sizes; (void)n_in; (void)out_size; (void)d_ws; (void)ws_size;
  const float* x1 = (const float*)d_in[0];
  const float* x2 = (const float*)d_in[1];
  const float* v  = (const float*)d_in[2];
  const float* h  = (const float*)d_in[3];
  const float* db = (const float*)d_in[4];
  float* out = (float*)d_out;
  dim3 grid(B_SIZE * D_MODEL);
  dim3 block(THREADS);
  hipLaunchKernelGGL(hyena_toeplitz_wmma, grid, block, 0, stream,
                     x1, x2, v, h, db, out);
}
